// GraphEncoder_43112881717724
// MI455X (gfx1250) — compile-verified, matrix operand order checked by reading, not run on hardware
//
#include <hip/hip_runtime.h>
#include <hip/hip_bf16.h>

typedef __attribute__((ext_vector_type(16))) __bf16       v16bf;
typedef __attribute__((ext_vector_type(8)))  float        v8f;
typedef __attribute__((ext_vector_type(4)))  unsigned int u32x4;
typedef __attribute__((ext_vector_type(8)))  int          i32x8;
typedef __attribute__((ext_vector_type(4)))  int          i32x4;

#define NNODES 50000
#define NEDGES 800000
#define NGRAPH 16
#define SEQ    512
#define D      128
#define NTOK   8
#define EMBED  4096
#define PROJ   (NTOK * EMBED)   /* 32768 */

/* ---------------- utility kernels ---------------- */

__global__ void k_zero(float* p, int n) {
    int i = blockIdx.x * blockDim.x + threadIdx.x;
    if (i < n) p[i] = 0.f;
}

__global__ void k_degree(const int* __restrict__ dst, float* __restrict__ cnt, int e) {
    int i = blockIdx.x * blockDim.x + threadIdx.x;
    if (i < e) atomicAdd(&cnt[dst[i]], 1.0f);
}

__global__ void k_inv(const float* __restrict__ cnt, float* __restrict__ inv, int n) {
    int i = blockIdx.x * blockDim.x + threadIdx.x;
    if (i < n) inv[i] = 1.0f / fmaxf(cnt[i], 1.0f);
}

/* pack a 128x128 f32 weight (K-major rows, N cols) into WMMA bf16 B-fragment order:
   dst[((kt*8 + nt)*32 + lane)*16 + e]  */
__global__ void k_prep_w128(const float* __restrict__ W, __bf16* __restrict__ Wf) {
    int t = blockIdx.x * blockDim.x + threadIdx.x;           // 16384 threads
    if (t >= 4 * 8 * 32 * 16) return;
    int e    = t & 15;
    int lane = (t >> 4) & 31;
    int nt   = (t >> 9) & 7;
    int kt   = t >> 12;
    int n    = nt * 16 + (lane & 15);
    int half = lane >> 4;
    int k    = kt * 32 + ((e < 8) ? (half * 8 + e) : (16 + half * 8 + (e - 8)));
    Wf[t] = (__bf16)W[k * D + n];
}

/* pack Wp (128 x 32768) -> dst[((nt*4 + kt)*32 + lane)*16 + e] */
__global__ void k_prep_wp(const float* __restrict__ W, __bf16* __restrict__ Wf) {
    int t = blockIdx.x * blockDim.x + threadIdx.x;           // 4194304 threads
    if (t >= D * PROJ) return;
    int e    = t & 15;
    int lane = (t >> 4) & 31;
    int kt   = (t >> 9) & 3;
    int nt   = t >> 11;
    int n    = nt * 16 + (lane & 15);
    int half = lane >> 4;
    int k    = kt * 32 + ((e < 8) ? (half * 8 + e) : (16 + half * 8 + (e - 8)));
    Wf[t] = (__bf16)W[(long)k * PROJ + n];
}

/* edge scatter: agg[dst] += h[src], 4 floats per thread */
__global__ void k_scatter_edges(const float* __restrict__ h, const int* __restrict__ srcE,
                                const int* __restrict__ dstE, float* __restrict__ agg) {
    long i = (long)blockIdx.x * blockDim.x + threadIdx.x;
    if (i >= (long)NEDGES * 32) return;
    int e = (int)(i >> 5);
    int c = ((int)i & 31) << 2;
    const float4 v = *(const float4*)(h + (long)srcE[e] * D + c);
    float* a = agg + (long)dstE[e] * D + c;
    atomicAdd(a + 0, v.x); atomicAdd(a + 1, v.y);
    atomicAdd(a + 2, v.z); atomicAdd(a + 3, v.w);
}

/* ---------------- TDM: async copy global -> LDS ----------------
   D# built per CDNA5 ISA:
   group0: count=1, lds_addr, global_addr, type=2
   group1: data_size=8B, tensor_dim0=tile_dim0=nbytes/8, dim1=1, stride0=nbytes/8
   This toolchain exposes the 6-arg builtin:
   (uint32x4 g0, int32x8 g1, int32x4 g2, int32x4 g3, int32x8 extra, i32 cpol) */
__device__ __forceinline__ void tdm_load_to_lds(const void* gsrc, unsigned lds_byte_addr,
                                                unsigned nbytes) {
    unsigned n8 = nbytes >> 3;                       // 8-byte elements (< 65536)
    unsigned long long ga = (unsigned long long)(uintptr_t)gsrc;
    u32x4 g0;
    g0[0] = 1u;                                       // count=1, user mode
    g0[1] = lds_byte_addr;                            // lds_addr (bits 63:32)
    g0[2] = (unsigned)(ga & 0xffffffffu);             // global_addr[31:0]
    g0[3] = (unsigned)((ga >> 32) & 0x01ffffffu)      // global_addr[56:32]
          | (2u << 30);                               // type=2 ("image")
    i32x8 g1;
    g1[0] = (int)(3u << 16);                          // data_size=3 (8 bytes)
    g1[1] = (int)(n8 << 16);                          // tensor_dim0[15:0]   (bits 63:48)
    g1[2] = (int)((n8 >> 16) | (1u << 16));           // tensor_dim0[31:16], tensor_dim1=1
    g1[3] = (int)(n8 << 16);                          // tile_dim0           (bits 127:112)
    g1[4] = 1;                                        // tile_dim1=1, tile_dim2=0
    g1[5] = (int)n8;                                  // tensor_dim0_stride[31:0]
    g1[6] = 0;
    g1[7] = 0;
    i32x4 z4 = {0, 0, 0, 0};                          // groups 2/3 unused (<=2D)
    i32x8 z8 = {0, 0, 0, 0, 0, 0, 0, 0};
    __builtin_amdgcn_tensor_load_to_lds(g0, g1, z4, z4, z8, 0);
}

/* ---------------- WMMA GEMM kernels ---------------- */

/* load one 16x32 bf16 A-fragment for this lane from an f32 row, scaling by s.
   lane holds row M=(lane&15); half=lane>>4 selects K sub-runs per ISA layout. */
__device__ __forceinline__ v16bf load_a_frag(const float* __restrict__ rowp, int kt, int half, float s) {
    int k0 = kt * 32 + half * 8;
    const float4* p = (const float4*)(rowp + k0);
    float4 x0 = p[0], x1 = p[1];
    const float4* q = (const float4*)(rowp + k0 + 16);
    float4 y0 = q[0], y1 = q[1];
    v16bf a;
    a[0]  = (__bf16)(x0.x * s); a[1]  = (__bf16)(x0.y * s);
    a[2]  = (__bf16)(x0.z * s); a[3]  = (__bf16)(x0.w * s);
    a[4]  = (__bf16)(x1.x * s); a[5]  = (__bf16)(x1.y * s);
    a[6]  = (__bf16)(x1.z * s); a[7]  = (__bf16)(x1.w * s);
    a[8]  = (__bf16)(y0.x * s); a[9]  = (__bf16)(y0.y * s);
    a[10] = (__bf16)(y0.z * s); a[11] = (__bf16)(y0.w * s);
    a[12] = (__bf16)(y1.x * s); a[13] = (__bf16)(y1.y * s);
    a[14] = (__bf16)(y1.z * s); a[15] = (__bf16)(y1.w * s);
    return a;
}

/* out = relu?( (agg*inv) @ Wl + h @ Wr + b )
   One wave per 16-row node block; the block's 8 waves share the packed
   Wl/Wr fragments, staged once into LDS by the Tensor Data Mover. */
__global__ __launch_bounds__(256)
void k_sage_gemm(const float* __restrict__ agg, const float* __restrict__ inv,
                 const float* __restrict__ h,
                 const __bf16* __restrict__ WlF, const __bf16* __restrict__ WrF,
                 const float* __restrict__ bias, float* __restrict__ out, int relu) {
    extern __shared__ __bf16 shW[];                  // 2 x 16384 bf16 = 64 KB
    int lane = threadIdx.x & 31;
    int wave = threadIdx.x >> 5;

    /* TDM stage: wave 0 DMAs both fragment blobs into LDS, waits TENSORcnt */
    if (wave == 0) {
        unsigned lbase = (unsigned)(unsigned long long)(uintptr_t)(void*)shW;
        tdm_load_to_lds(WlF, lbase,          16384u * 2u);
        tdm_load_to_lds(WrF, lbase + 32768u, 16384u * 2u);
        __builtin_amdgcn_s_wait_tensorcnt(0);
    }
    __syncthreads();                                 // publish LDS to all waves

    int rb = blockIdx.x * 8 + wave;
    bool active = (rb < NNODES / 16);                // wave-uniform: EXEC stays all-ones
    if (active) {
        int mrow = rb * 16 + (lane & 15);
        int half = lane >> 4;
        int ncol = lane & 15;
        float s = inv[mrow];
        const float* aggRow = agg + (long)mrow * D;
        const float* hRow   = h   + (long)mrow * D;
        const __bf16* shWl = shW;
        const __bf16* shWr = shW + 16384;

        v8f acc[8];
#pragma unroll
        for (int nt = 0; nt < 8; ++nt) {
            float bv = bias[nt * 16 + ncol];
#pragma unroll
            for (int r = 0; r < 8; ++r) acc[nt][r] = bv;
        }
#pragma unroll
        for (int kt = 0; kt < 4; ++kt) {
            v16bf aA = load_a_frag(aggRow, kt, half, s);
            v16bf aH = load_a_frag(hRow,   kt, half, 1.0f);
#pragma unroll
            for (int nt = 0; nt < 8; ++nt) {
                v16bf bl = *(const v16bf*)(shWl + ((kt * 8 + nt) * 32 + lane) * 16);
                v16bf br = *(const v16bf*)(shWr + ((kt * 8 + nt) * 32 + lane) * 16);
                acc[nt] = __builtin_amdgcn_wmma_f32_16x16x32_bf16(false, aA, false, bl,
                                                                  (short)0, acc[nt], false, false);
                acc[nt] = __builtin_amdgcn_wmma_f32_16x16x32_bf16(false, aH, false, br,
                                                                  (short)0, acc[nt], false, false);
            }
        }
        int mbase = rb * 16 + 8 * half;              // C layout: lanes 0-15 -> M=r, 16-31 -> M=8+r
#pragma unroll
        for (int nt = 0; nt < 8; ++nt) {
            int n = nt * 16 + ncol;
#pragma unroll
            for (int r = 0; r < 8; ++r) {
                float v = acc[nt][r];
                if (relu) v = fmaxf(v, 0.f);
                out[(long)(mbase + r) * D + n] = v;
            }
        }
    }
}

/* node_embedding[16 x 32768] = pooled @ Wp + bp ; one wave per 16-wide col tile */
__global__ __launch_bounds__(256)
void k_proj_gemm(const float* __restrict__ pooled, const __bf16* __restrict__ WpF,
                 const float* __restrict__ bp, float* __restrict__ outE) {
    int lane = threadIdx.x & 31;
    int wave = threadIdx.x >> 5;
    int nt   = blockIdx.x * 8 + wave;
    if (nt >= PROJ / 16) return;
    int half = lane >> 4;
    int ncol = lane & 15;
    const float* rowp = pooled + (lane & 15) * D;
    float bv = bp[nt * 16 + ncol];
    v8f acc;
#pragma unroll
    for (int r = 0; r < 8; ++r) acc[r] = bv;
#pragma unroll
    for (int kt = 0; kt < 4; ++kt) {
        v16bf a = load_a_frag(rowp, kt, half, 1.0f);
        v16bf b = *(const v16bf*)(WpF + ((nt * 4 + kt) * 32 + lane) * 16);
        acc = __builtin_amdgcn_wmma_f32_16x16x32_bf16(false, a, false, b,
                                                      (short)0, acc, false, false);
    }
    int mbase = 8 * half;
#pragma unroll
    for (int r = 0; r < 8; ++r)
        outE[(long)(mbase + r) * PROJ + nt * 16 + ncol] = acc[r];
}

/* ---------------- pooling + assembly ---------------- */

__global__ void k_pool_scatter(const float* __restrict__ h, const int* __restrict__ batch,
                               float* __restrict__ sums, float* __restrict__ cntB) {
    long i = (long)blockIdx.x * blockDim.x + threadIdx.x;
    if (i >= (long)NNODES * 32) return;
    int node = (int)(i >> 5);
    int c = ((int)i & 31) << 2;
    int g = batch[node];
    const float4 v = *(const float4*)(h + (long)node * D + c);
    float* sp = sums + g * D + c;
    atomicAdd(sp + 0, v.x); atomicAdd(sp + 1, v.y);
    atomicAdd(sp + 2, v.z); atomicAdd(sp + 3, v.w);
    if (((int)i & 31) == 0) atomicAdd(&cntB[g], 1.0f);
}

__global__ void k_pool_finish(const float* __restrict__ sums, const float* __restrict__ cntB,
                              float* __restrict__ pooled) {
    int i = blockIdx.x * blockDim.x + threadIdx.x;
    if (i < NGRAPH * D) pooled[i] = sums[i] / fmaxf(cntB[i / D], 1.0f);
}

/* out[b,s,:] = s<8 ? node_embedding[b, s*4096 ...] : embed_tokens[ids[b,s]] */
__global__ void k_assemble(const float* __restrict__ embT, const int* __restrict__ ids,
                           const float* __restrict__ nodeE, float* __restrict__ out) {
    int row = blockIdx.x;                        // 0 .. B*S-1
    int s = row & (SEQ - 1);
    int b = row >> 9;
    const float* srcp = (s < NTOK) ? (nodeE + (long)b * PROJ + (long)s * EMBED)
                                   : (embT + (long)ids[row] * EMBED);
    const float4* s4 = (const float4*)srcp;
    float4* o4 = (float4*)(out + (long)row * EMBED);
    int t = threadIdx.x;
#pragma unroll
    for (int j = 0; j < 4; ++j) o4[t + j * 256] = s4[t + j * 256];
}

/* ---------------- host launcher ---------------- */

extern "C" void kernel_launch(void* const* d_in, const int* in_sizes, int n_in,
                              void* d_out, int out_size, void* d_ws, size_t ws_size,
                              hipStream_t stream) {
    const float* x     = (const float*)d_in[0];
    const int*   edge  = (const int*)d_in[1];
    const int*   srcE  = edge;
    const int*   dstE  = edge + NEDGES;
    const int*   batch = (const int*)d_in[2];
    const int*   ids   = (const int*)d_in[3];
    /* d_in[4] = is_node: by construction exactly first NTOK per row; not needed */
    const float* embT  = (const float*)d_in[5];
    const float* Wl0 = (const float*)d_in[6],  *Wr0 = (const float*)d_in[7],  *b0 = (const float*)d_in[8];
    const float* Wl1 = (const float*)d_in[9],  *Wr1 = (const float*)d_in[10], *b1 = (const float*)d_in[11];
    const float* Wl2 = (const float*)d_in[12], *Wr2 = (const float*)d_in[13], *b2 = (const float*)d_in[14];
    const float* Wp  = (const float*)d_in[15], *bp  = (const float*)d_in[16];

    char* ws = (char*)d_ws;
    size_t off = 0;
    auto take = [&](size_t bytes) { size_t r = off; off += (bytes + 255) & ~(size_t)255; return r; };
    float*  agg   = (float*)(ws + take((size_t)NNODES * D * 4));
    float*  hA    = (float*)(ws + take((size_t)NNODES * D * 4));
    float*  hB    = (float*)(ws + take((size_t)NNODES * D * 4));
    float*  cnt   = (float*)(ws + take((size_t)NNODES * 4));
    float*  inv   = (float*)(ws + take((size_t)NNODES * 4));
    __bf16* wf    = (__bf16*)(ws + take((size_t)6 * D * D * 2));
    __bf16* wpf   = (__bf16*)(ws + take((size_t)D * PROJ * 2));
    float*  sums  = (float*)(ws + take((size_t)NGRAPH * D * 4));
    float*  cntB  = (float*)(ws + take((size_t)NGRAPH * 4));
    float*  pooled= (float*)(ws + take((size_t)NGRAPH * D * 4));
    float*  nodeE = (float*)(ws + take((size_t)NGRAPH * PROJ * 4));

    const int T = 256;
    const size_t SHW = 2u * 16384u * 2u;             // 64 KB dynamic LDS for k_sage_gemm

    /* degree + inverse (shared by all layers) */
    k_zero<<<(NNODES + T - 1) / T, T, 0, stream>>>(cnt, NNODES);
    k_degree<<<(NEDGES + T - 1) / T, T, 0, stream>>>(dstE, cnt, NEDGES);
    k_inv<<<(NNODES + T - 1) / T, T, 0, stream>>>(cnt, inv, NNODES);

    /* weight packing to WMMA fragment order */
    k_prep_w128<<<64, T, 0, stream>>>(Wl0, wf + 0 * 16384);
    k_prep_w128<<<64, T, 0, stream>>>(Wr0, wf + 1 * 16384);
    k_prep_w128<<<64, T, 0, stream>>>(Wl1, wf + 2 * 16384);
    k_prep_w128<<<64, T, 0, stream>>>(Wr1, wf + 3 * 16384);
    k_prep_w128<<<64, T, 0, stream>>>(Wl2, wf + 4 * 16384);
    k_prep_w128<<<64, T, 0, stream>>>(Wr2, wf + 5 * 16384);
    k_prep_wp<<<(D * PROJ + T - 1) / T, T, 0, stream>>>(Wp, wpf);

    const int gemmGrid = (NNODES / 16 + 7) / 8;      // 391 blocks, 8 waves each
    const int zeroND   = (NNODES * D + T - 1) / T;
    const int scatGrid = (int)(((long)NEDGES * 32 + T - 1) / T);

    /* layer 0 */
    k_zero<<<zeroND, T, 0, stream>>>(agg, NNODES * D);
    k_scatter_edges<<<scatGrid, T, 0, stream>>>(x, srcE, dstE, agg);
    k_sage_gemm<<<gemmGrid, T, SHW, stream>>>(agg, inv, x, wf + 0 * 16384, wf + 1 * 16384, b0, hA, 1);
    /* layer 1 */
    k_zero<<<zeroND, T, 0, stream>>>(agg, NNODES * D);
    k_scatter_edges<<<scatGrid, T, 0, stream>>>(hA, srcE, dstE, agg);
    k_sage_gemm<<<gemmGrid, T, SHW, stream>>>(agg, inv, hA, wf + 2 * 16384, wf + 3 * 16384, b1, hB, 1);
    /* layer 2 (no relu) */
    k_zero<<<zeroND, T, 0, stream>>>(agg, NNODES * D);
    k_scatter_edges<<<scatGrid, T, 0, stream>>>(hB, srcE, dstE, agg);
    k_sage_gemm<<<gemmGrid, T, SHW, stream>>>(agg, inv, hB, wf + 4 * 16384, wf + 5 * 16384, b2, hA, 0);

    /* mean pool over graphs */
    k_zero<<<(NGRAPH * D + T - 1) / T, T, 0, stream>>>(sums, NGRAPH * D);
    k_zero<<<1, T, 0, stream>>>(cntB, NGRAPH);
    k_pool_scatter<<<(int)(((long)NNODES * 32 + T - 1) / T), T, 0, stream>>>(hA, batch, sums, cntB);
    k_pool_finish<<<(NGRAPH * D + T - 1) / T, T, 0, stream>>>(sums, cntB, pooled);

    /* projection to graph tokens */
    k_proj_gemm<<<(PROJ / 16 + 7) / 8, T, 0, stream>>>(pooled, wpf, bp, nodeE);

    /* final embedding assembly */
    k_assemble<<<NGRAPH * SEQ, T, 0, stream>>>(embT, ids, nodeE, (float*)d_out);
}
